// GGNN_37709812859114
// MI455X (gfx1250) — matrix operand order repaired and checked by reading, not compiled, not run on hardware
//
#include <hip/hip_runtime.h>

#define NNODES  8000
#define NEDGES  256000
#define SD      64
#define K1      131
#define K1P     160
#define NSTEPS  10

typedef __attribute__((ext_vector_type(16))) __bf16 v16bf;
typedef __attribute__((ext_vector_type(8)))  __bf16 v8bf;
typedef __attribute__((ext_vector_type(2)))  __bf16 v2bf;
typedef __attribute__((ext_vector_type(8)))  float  v8f;

static __device__ __forceinline__ __bf16 f2bf(float f) {
  unsigned u = __builtin_bit_cast(unsigned, f);
  u += 0x7FFFu + ((u >> 16) & 1u);           // round-to-nearest-even
  unsigned short h = (unsigned short)(u >> 16);
  return __builtin_bit_cast(__bf16, h);
}
static __device__ __forceinline__ float bf2f(__bf16 b) {
  unsigned short s = __builtin_bit_cast(unsigned short, b);
  unsigned u = ((unsigned)s) << 16;
  return __builtin_bit_cast(float, u);
}
// packed f32x2 -> bf16x2 (one v_cvt_pk_bf16_f32 when available)
static __device__ __forceinline__ v2bf pk2(float a, float b) {
#if __has_builtin(__builtin_amdgcn_cvt_pk_bf16_f32)
  return __builtin_amdgcn_cvt_pk_bf16_f32(a, b);
#else
  v2bf r; r[0] = f2bf(a); r[1] = f2bf(b); return r;
#endif
}
static __device__ __forceinline__ unsigned pk2u(float a, float b) {
  return __builtin_bit_cast(unsigned, pk2(a, b));
}
static __device__ __forceinline__ v16bf cat8(v8bf lo, v8bf hi) {
  return __builtin_shufflevector(lo, hi, 0,1,2,3,4,5,6,7,8,9,10,11,12,13,14,15);
}
// A fragment (16x32 bf16): lane half K-offset 0/8, elems {base..base+7, base+16..base+23}
static __device__ __forceinline__ v16bf ldA(const __bf16* p) {
  v8bf lo = *(const v8bf*)(p);
  v8bf hi = *(const v8bf*)(p + 16);
  return cat8(lo, hi);
}
// B fragment (32x16 bf16): lane half K-offset 0/8, elems {base..base+7, base+16..base+23}
static __device__ __forceinline__ v16bf ldB(const __bf16* p) {
  v8bf lo = *(const v8bf*)(p);
  v8bf hi = *(const v8bf*)(p + 16);
  return cat8(lo, hi);
}
#define WMMA_BF16(A,B,C) \
  __builtin_amdgcn_wmma_f32_16x16x32_bf16(false,(A),false,(B),(short)0,(C),false,false)

// ---------------------------------------------------------------------------
// per-edge scalar precompute: Jv = J[row,col], br = b[row], bc = b[col]
// ---------------------------------------------------------------------------
__global__ __launch_bounds__(256) void edge_scalars_k(
    const float* __restrict__ J, const float* __restrict__ b,
    const int* __restrict__ row, const int* __restrict__ col,
    float* __restrict__ Jv, float* __restrict__ brr, float* __restrict__ bcc)
{
  int e = blockIdx.x * blockDim.x + threadIdx.x;
  if (e < NEDGES) {
    int r = row[e], c = col[e];
    Jv[e]  = J[(size_t)r * NNODES + c];
    brr[e] = b[r];
    bcc[e] = b[c];
  }
}

// ---------------------------------------------------------------------------
// edge MLP (131 -> 128 -> 128 -> 64) + scatter-add onto nm[row]
// ---------------------------------------------------------------------------
__global__ __launch_bounds__(256) void edge_mlp_scatter_k(
    const float* __restrict__ h,
    const float* __restrict__ Jv, const float* __restrict__ brr, const float* __restrict__ bcc,
    const int* __restrict__ row, const int* __restrict__ col,
    const float* __restrict__ Wm1, const float* __restrict__ bm1,
    const float* __restrict__ Wm2, const float* __restrict__ bm2,
    const float* __restrict__ Wm3, const float* __restrict__ bm3,
    float* __restrict__ nm)
{
  extern __shared__ __align__(16) char smem[];
  float*  bm1f = (float*)smem;                 // 128
  float*  bm2f = bm1f + 128;                   // 128
  float*  bm3f = bm2f + 128;                   // 64   (1280 B total)
  __bf16* w1   = (__bf16*)(bm3f + 64);         // [128][160]
  __bf16* w2   = w1 + 128 * K1P;               // [128][128]
  __bf16* w3   = w2 + 128 * 128;               // [64][128]
  __bf16* emal = w3 + 64 * 128;                // 8 waves x [16][160]

  const int tid = threadIdx.x;
  for (int i = tid; i < 128 * K1P; i += 256) {
    int n = i / K1P, k = i % K1P;
    w1[i] = (k < K1) ? f2bf(Wm1[n * K1 + k]) : f2bf(0.f);
  }
  for (int i = tid; i < 128 * 128; i += 256) w2[i] = f2bf(Wm2[i]);
  for (int i = tid; i < 64 * 128;  i += 256) w3[i] = f2bf(Wm3[i]);
  for (int i = tid; i < 128; i += 256) { bm1f[i] = bm1[i]; bm2f[i] = bm2[i]; }
  for (int i = tid; i < 64;  i += 256) bm3f[i] = bm3[i];
  __syncthreads();

  const int wave = tid >> 5, lane = tid & 31;
  __bf16* em = emal + wave * (16 * K1P);
  const int m   = lane & 15;
  const int hiA = (lane >> 4) << 3;   // 0 / 8  (A-frag K offset, C/D M offset)
  const int hiB = (lane >> 4) << 3;   // 0 / 8  (B-frag K offset)

  const int nTiles = NEDGES / 16;
  for (int tile = blockIdx.x * 8 + wave; tile < nTiles; tile += (int)gridDim.x * 8) {
    const int e0 = tile * 16;
    { // stage [16 x 160] bf16 edge features: [h[row] | h[col] | Jv br bc | 0-pad]
      const int mm = lane >> 1, half = lane & 1;
      const int e = e0 + mm;
      const int r = row[e], c = col[e];
      const float4* hr = (const float4*)(h + (size_t)r * SD + half * 32);
      const float4* hc = (const float4*)(h + (size_t)c * SD + half * 32);
      unsigned* d = (unsigned*)(em + mm * K1P + half * 32);
#pragma unroll
      for (int q = 0; q < 8; ++q) {
        float4 vr = hr[q], vc = hc[q];
        d[q*2+0]      = pk2u(vr.x, vr.y);
        d[q*2+1]      = pk2u(vr.z, vr.w);
        d[32 + q*2+0] = pk2u(vc.x, vc.y);
        d[32 + q*2+1] = pk2u(vc.z, vc.w);
      }
      unsigned* rowd = (unsigned*)(em + mm * K1P);
      if (half == 0) {
        rowd[64] = pk2u(Jv[e], brr[e]);     // elems 128,129
        rowd[65] = pk2u(bcc[e], 0.f);       // elems 130,131
#pragma unroll
        for (int w = 66; w < 72; ++w) rowd[w] = 0u;   // elems 132..143
      } else {
#pragma unroll
        for (int w = 72; w < 80; ++w) rowd[w] = 0u;   // elems 144..159
      }
    }
    // ----- layer 1: [16x160] @ [160x128] -----
    v8f acc[8] = {};
#pragma unroll
    for (int kt = 0; kt < 5; ++kt) {
      v16bf a = ldA(em + m * K1P + kt * 32 + hiA);
#pragma unroll
      for (int nt = 0; nt < 8; ++nt) {
        v16bf b = ldB(w1 + (nt * 16 + m) * K1P + kt * 32 + hiB);
        acc[nt] = WMMA_BF16(a, b, acc[nt]);
      }
    }
    __bf16* act = em;  // reuse staging buffer, stride 128
#pragma unroll
    for (int nt = 0; nt < 8; ++nt) {
      const int c = nt * 16 + m; const float bia = bm1f[c];
#pragma unroll
      for (int i = 0; i < 8; i += 2) {
        float v0 = acc[nt][i]   + bia; v0 = v0 > 0.f ? v0 : 0.f;
        float v1 = acc[nt][i+1] + bia; v1 = v1 > 0.f ? v1 : 0.f;
        v2bf r = pk2(v0, v1);
        act[(i     + hiA) * 128 + c] = r[0];
        act[(i + 1 + hiA) * 128 + c] = r[1];
      }
    }
    // ----- layer 2: [16x128] @ [128x128] -----
    v8f acc2[8] = {};
#pragma unroll
    for (int kt = 0; kt < 4; ++kt) {
      v16bf a = ldA(act + m * 128 + kt * 32 + hiA);
#pragma unroll
      for (int nt = 0; nt < 8; ++nt) {
        v16bf b = ldB(w2 + (nt * 16 + m) * 128 + kt * 32 + hiB);
        acc2[nt] = WMMA_BF16(a, b, acc2[nt]);
      }
    }
#pragma unroll
    for (int nt = 0; nt < 8; ++nt) {
      const int c = nt * 16 + m; const float bia = bm2f[c];
#pragma unroll
      for (int i = 0; i < 8; i += 2) {
        float v0 = acc2[nt][i]   + bia; v0 = v0 > 0.f ? v0 : 0.f;
        float v1 = acc2[nt][i+1] + bia; v1 = v1 > 0.f ? v1 : 0.f;
        v2bf r = pk2(v0, v1);
        act[(i     + hiA) * 128 + c] = r[0];
        act[(i + 1 + hiA) * 128 + c] = r[1];
      }
    }
    // ----- layer 3: [16x128] @ [128x64] + scatter-add -----
    v8f acc3[4] = {};
#pragma unroll
    for (int kt = 0; kt < 4; ++kt) {
      v16bf a = ldA(act + m * 128 + kt * 32 + hiA);
#pragma unroll
      for (int nt = 0; nt < 4; ++nt) {
        v16bf b = ldB(w3 + (nt * 16 + m) * 128 + kt * 32 + hiB);
        acc3[nt] = WMMA_BF16(a, b, acc3[nt]);
      }
    }
#pragma unroll
    for (int nt = 0; nt < 4; ++nt) {
      const int c = nt * 16 + m; const float bia = bm3f[c];
#pragma unroll
      for (int i = 0; i < 8; ++i) {
        const int e = e0 + i + hiA;
        const int r = row[e];
        unsafeAtomicAdd(&nm[(size_t)r * SD + c], acc3[nt][i] + bia);
      }
    }
  }
}

// ---------------------------------------------------------------------------
// GRU cell: h' = (1-z)*tanh(i_n + r*h_n) + z*h
// ---------------------------------------------------------------------------
__global__ __launch_bounds__(256) void gru_update_k(
    const float* __restrict__ nm, const float* __restrict__ hcur,
    const float* __restrict__ W_ih, const float* __restrict__ W_hh,
    const float* __restrict__ b_ih, const float* __restrict__ b_hh,
    float* __restrict__ hnext)
{
  extern __shared__ __align__(16) char smem[];
  float*  bihf = (float*)smem;            // 192
  float*  bhhf = bihf + 192;              // 192 (1536 B)
  __bf16* wih  = (__bf16*)(bhhf + 192);   // [192][64]
  __bf16* whh  = wih + 192 * 64;          // [192][64]
  __bf16* stg  = whh + 192 * 64;          // 8 waves x (1024 nm + 1024 h) bf16

  const int tid = threadIdx.x;
  for (int i = tid; i < 192 * 64; i += 256) { wih[i] = f2bf(W_ih[i]); whh[i] = f2bf(W_hh[i]); }
  for (int i = tid; i < 192; i += 256) { bihf[i] = b_ih[i]; bhhf[i] = b_hh[i]; }
  __syncthreads();

  const int wave = tid >> 5, lane = tid & 31;
  __bf16* nmA = stg + wave * 2048;
  __bf16* hA  = nmA + 1024;
  const int m = lane & 15;
  const int hiA = (lane >> 4) << 3;
  const int hiB = (lane >> 4) << 3;

  const int nTiles = NNODES / 16;
  for (int tile = blockIdx.x * 8 + wave; tile < nTiles; tile += (int)gridDim.x * 8) {
    const int n0 = tile * 16;
    { // stage nm / h tiles (bf16, packed 32-bit stores)
      const int mm = lane >> 1, half = lane & 1;
      const int node = n0 + mm;
      const float4* pn = (const float4*)(nm   + (size_t)node * SD + half * 32);
      const float4* ph = (const float4*)(hcur + (size_t)node * SD + half * 32);
      unsigned* dn = (unsigned*)(nmA + mm * 64 + half * 32);
      unsigned* dh = (unsigned*)(hA  + mm * 64 + half * 32);
#pragma unroll
      for (int q = 0; q < 8; ++q) {
        float4 a = pn[q], b = ph[q];
        dn[q*2+0] = pk2u(a.x, a.y); dn[q*2+1] = pk2u(a.z, a.w);
        dh[q*2+0] = pk2u(b.x, b.y); dh[q*2+1] = pk2u(b.z, b.w);
      }
    }
#pragma unroll
    for (int ct = 0; ct < 4; ++ct) {      // 4 x 16 columns of the 64-d state
      v8f gi[3] = {}; v8f gh[3] = {};
#pragma unroll
      for (int kt = 0; kt < 2; ++kt) {
        v16bf an = ldA(nmA + m * 64 + kt * 32 + hiA);
        v16bf ah = ldA(hA  + m * 64 + kt * 32 + hiA);
#pragma unroll
        for (int g = 0; g < 3; ++g) {     // r, z, n gates
          const int cb = g * 64 + ct * 16 + m;
          v16bf bi = ldB(wih + cb * 64 + kt * 32 + hiB);
          v16bf bh = ldB(whh + cb * 64 + kt * 32 + hiB);
          gi[g] = WMMA_BF16(an, bi, gi[g]);
          gh[g] = WMMA_BF16(ah, bh, gh[g]);
        }
      }
      const int c = ct * 16 + m;
      const float bir = bihf[c],       bhr = bhhf[c];
      const float biz = bihf[64 + c],  bhz = bhhf[64 + c];
      const float bin = bihf[128 + c], bhn = bhhf[128 + c];
#pragma unroll
      for (int i = 0; i < 8; ++i) {
        const int node = n0 + i + hiA;
        float r  = 1.f / (1.f + __expf(-((gi[0][i] + bir) + (gh[0][i] + bhr))));
        float z  = 1.f / (1.f + __expf(-((gi[1][i] + biz) + (gh[1][i] + bhz))));
        float ng = tanhf((gi[2][i] + bin) + r * (gh[2][i] + bhn));
        float ho = hcur[(size_t)node * SD + c];
        hnext[(size_t)node * SD + c] = (1.f - z) * ng + z * ho;
      }
    }
  }
}

// ---------------------------------------------------------------------------
// readout: relu(h@Wr1')->relu(@Wr2')->@Wr3' -> (softmax, logits)
// ---------------------------------------------------------------------------
__global__ __launch_bounds__(256) void readout_k(
    const float* __restrict__ h,
    const float* __restrict__ Wr1, const float* __restrict__ br1,
    const float* __restrict__ Wr2, const float* __restrict__ br2,
    const float* __restrict__ Wr3, const float* __restrict__ br3,
    float* __restrict__ out)
{
  extern __shared__ __align__(16) char smem[];
  float*  br1f = (float*)smem;            // 128
  float*  br2f = br1f + 128;              // 128
  float*  wr3f = br2f + 128;              // 256  ([2][128] f32)
  float*  br3f = wr3f + 256;              // 4 (2 used) -> 2064 B total
  __bf16* w1   = (__bf16*)(br3f + 4);     // [128][64]
  __bf16* w2   = w1 + 128 * 64;           // [128][128]
  __bf16* stg  = w2 + 128 * 128;          // 8 waves x (1024 hbuf + 2048 act)

  const int tid = threadIdx.x;
  for (int i = tid; i < 128 * 64;  i += 256) w1[i] = f2bf(Wr1[i]);
  for (int i = tid; i < 128 * 128; i += 256) w2[i] = f2bf(Wr2[i]);
  for (int i = tid; i < 128; i += 256) { br1f[i] = br1[i]; br2f[i] = br2[i]; }
  for (int i = tid; i < 256; i += 256) wr3f[i] = Wr3[i];
  if (tid < 2) br3f[tid] = br3[tid];
  __syncthreads();

  const int wave = tid >> 5, lane = tid & 31;
  __bf16* hbuf = stg + wave * 3072;
  __bf16* act  = hbuf + 1024;
  const int m = lane & 15;
  const int hiA = (lane >> 4) << 3;
  const int hiB = (lane >> 4) << 3;

  const int nTiles = NNODES / 16;
  for (int tile = blockIdx.x * 8 + wave; tile < nTiles; tile += (int)gridDim.x * 8) {
    const int n0 = tile * 16;
    { // stage h tile
      const int mm = lane >> 1, half = lane & 1;
      const float4* ph = (const float4*)(h + (size_t)(n0 + mm) * SD + half * 32);
      unsigned* dh = (unsigned*)(hbuf + mm * 64 + half * 32);
#pragma unroll
      for (int q = 0; q < 8; ++q) {
        float4 b = ph[q];
        dh[q*2+0] = pk2u(b.x, b.y); dh[q*2+1] = pk2u(b.z, b.w);
      }
    }
    // L1: [16x64] @ [64x128]
    v8f acc[8] = {};
#pragma unroll
    for (int kt = 0; kt < 2; ++kt) {
      v16bf a = ldA(hbuf + m * 64 + kt * 32 + hiA);
#pragma unroll
      for (int nt = 0; nt < 8; ++nt) {
        v16bf b = ldB(w1 + (nt * 16 + m) * 64 + kt * 32 + hiB);
        acc[nt] = WMMA_BF16(a, b, acc[nt]);
      }
    }
#pragma unroll
    for (int nt = 0; nt < 8; ++nt) {
      const int c = nt * 16 + m; const float bia = br1f[c];
#pragma unroll
      for (int i = 0; i < 8; i += 2) {
        float v0 = acc[nt][i]   + bia; v0 = v0 > 0.f ? v0 : 0.f;
        float v1 = acc[nt][i+1] + bia; v1 = v1 > 0.f ? v1 : 0.f;
        v2bf r = pk2(v0, v1);
        act[(i     + hiA) * 128 + c] = r[0];
        act[(i + 1 + hiA) * 128 + c] = r[1];
      }
    }
    // L2: [16x128] @ [128x128]
    v8f acc2[8] = {};
#pragma unroll
    for (int kt = 0; kt < 4; ++kt) {
      v16bf a = ldA(act + m * 128 + kt * 32 + hiA);
#pragma unroll
      for (int nt = 0; nt < 8; ++nt) {
        v16bf b = ldB(w2 + (nt * 16 + m) * 128 + kt * 32 + hiB);
        acc2[nt] = WMMA_BF16(a, b, acc2[nt]);
      }
    }
#pragma unroll
    for (int nt = 0; nt < 8; ++nt) {
      const int c = nt * 16 + m; const float bia = br2f[c];
#pragma unroll
      for (int i = 0; i < 8; i += 2) {
        float v0 = acc2[nt][i]   + bia; v0 = v0 > 0.f ? v0 : 0.f;
        float v1 = acc2[nt][i+1] + bia; v1 = v1 > 0.f ? v1 : 0.f;
        v2bf r = pk2(v0, v1);
        act[(i     + hiA) * 128 + c] = r[0];
        act[(i + 1 + hiA) * 128 + c] = r[1];
      }
    }
    // L3: [16x128] @ [128x2] + softmax over the 2 logits
    const int j = lane >> 4;              // lanes 0-15: class 0, 16-31: class 1
    float logit = br3f[j];
#pragma unroll 8
    for (int k = 0; k < 128; ++k) logit += bf2f(act[m * 128 + k]) * wr3f[j * 128 + k];
    float other = __shfl_xor(logit, 16, 32);
    float mx = fmaxf(logit, other);
    float ea = __expf(logit - mx), eb = __expf(other - mx);
    float sm = ea / (ea + eb);
    const int node = n0 + m;
    out[(size_t)node * 2 + j] = sm;                         // readout_final
    out[(size_t)NNODES * 2 + (size_t)node * 2 + j] = logit; // readout
  }
}

// ---------------------------------------------------------------------------
extern "C" void kernel_launch(void* const* d_in, const int* in_sizes, int n_in,
                              void* d_out, int out_size, void* d_ws, size_t ws_size,
                              hipStream_t stream)
{
  (void)in_sizes; (void)n_in; (void)out_size; (void)ws_size;
  const float* J    = (const float*)d_in[0];
  const float* b    = (const float*)d_in[1];
  const int*   row  = (const int*)  d_in[2];
  const int*   col  = (const int*)  d_in[3];
  const float* Wm1  = (const float*)d_in[4];
  const float* bm1  = (const float*)d_in[5];
  const float* Wm2  = (const float*)d_in[6];
  const float* bm2  = (const float*)d_in[7];
  const float* Wm3  = (const float*)d_in[8];
  const float* bm3  = (const float*)d_in[9];
  const float* W_ih = (const float*)d_in[10];
  const float* W_hh = (const float*)d_in[11];
  const float* b_ih = (const float*)d_in[12];
  const float* b_hh = (const float*)d_in[13];
  const float* Wr1  = (const float*)d_in[14];
  const float* br1  = (const float*)d_in[15];
  const float* Wr2  = (const float*)d_in[16];
  const float* br2  = (const float*)d_in[17];
  const float* Wr3  = (const float*)d_in[18];
  const float* br3  = (const float*)d_in[19];

  float* h0  = (float*)d_ws;
  float* h1  = h0 + (size_t)NNODES * SD;
  float* nm  = h1 + (size_t)NNODES * SD;
  float* Jv  = nm + (size_t)NNODES * SD;
  float* brr = Jv + NEDGES;
  float* bcc = brr + NEDGES;

  constexpr size_t EDGE_LDS = 1280 + 128*K1P*2 + 128*128*2 + 64*128*2 + 8*16*K1P*2; // 132352
  constexpr size_t GRU_LDS  = 1536 + 2*192*64*2 + 8*4096;                            // 83456
  constexpr size_t RO_LDS   = 2064 + 128*64*2 + 128*128*2 + 8*6144;                  // 100368

  hipMemsetAsync(h0, 0, (size_t)NNODES * SD * sizeof(float), stream);
  edge_scalars_k<<<(NEDGES + 255) / 256, 256, 0, stream>>>(J, b, row, col, Jv, brr, bcc);

  float* hc = h0; float* hn = h1;
  for (int s = 0; s < NSTEPS; ++s) {
    hipMemsetAsync(nm, 0, (size_t)NNODES * SD * sizeof(float), stream);
    edge_mlp_scatter_k<<<200, 256, EDGE_LDS, stream>>>(hc, Jv, brr, bcc, row, col,
                                                       Wm1, bm1, Wm2, bm2, Wm3, bm3, nm);
    gru_update_k<<<64, 256, GRU_LDS, stream>>>(nm, hc, W_ih, W_hh, b_ih, b_hh, hn);
    float* t = hc; hc = hn; hn = t;
  }
  readout_k<<<64, 256, RO_LDS, stream>>>(hc, Wr1, br1, Wr2, br2, Wr3, br3, (float*)d_out);
}
